// VSSBlock_15676630630815
// MI455X (gfx1250) — matrix operand order repaired and checked by reading, not compiled, not run on hardware
//
#include <hip/hip_runtime.h>
#include <hip/hip_bf16.h>
#include <stdint.h>

// ---------------------------------------------------------------- feature probes
#if defined(__has_builtin)
# if __has_builtin(__builtin_amdgcn_global_load_async_to_lds_b32) && \
     __has_builtin(__builtin_amdgcn_s_wait_asynccnt)
#  define HAVE_ASYNC_LDS 1
# endif
#endif
#ifndef HAVE_ASYNC_LDS
# define HAVE_ASYNC_LDS 0
#endif

// ---------------------------------------------------------------- constants
constexpr int Bn  = 4;
constexpr int HWn = 64 * 64;     // 4096
constexpr int Ln  = 4096;
constexpr int DIn = 192;
constexpr int Kd  = 4;           // scan directions
constexpr int PP  = 66 * 66;     // padded plane (H+2)*(W+2) = 4356

typedef __attribute__((ext_vector_type(16))) __bf16 bf16x16;
typedef __attribute__((ext_vector_type(8)))  float  f32x8;

__device__ __forceinline__ __bf16 f2bf(float f) { return static_cast<__bf16>(f); }

__device__ __forceinline__ void fill8(bf16x16& af, int base, float4 a, float4 b) {
  af[base + 0] = f2bf(a.x); af[base + 1] = f2bf(a.y);
  af[base + 2] = f2bf(a.z); af[base + 3] = f2bf(a.w);
  af[base + 4] = f2bf(b.x); af[base + 5] = f2bf(b.y);
  af[base + 6] = f2bf(b.z); af[base + 7] = f2bf(b.w);
}

// ---------------------------------------------------------------- weight pack
// Pack f32 weights [Nreal, Kreal] into WMMA B-fragment order:
//   dst[((kt*ntn + nt)*32 + lane)*16 + i]  holds (k = kt*32 + (i<8 ? (lane>=16?8:0) + i
//                                                : 16 + (lane>=16?8:0) + i-8), n = nt*16 + lane%16)
// K-order remap for convs: k = tap*Cin + ci maps to source k' = ci*taps + tap.
__global__ void k_pack(const float* __restrict__ src, __bf16* __restrict__ dst,
                       int Kreal, int Nreal, int Npad, int taps) {
  int ntn = Npad >> 4;
  long long tot = (long long)Kreal * Npad;
  long long idx = (long long)blockIdx.x * 256 + threadIdx.x;
  if (idx >= tot) return;
  int i    = (int)(idx & 15);
  int lane = (int)((idx >> 4) & 31);
  long long t2 = idx >> 9;
  int nt = (int)(t2 % ntn);
  int kt = (int)(t2 / ntn);
  int k = kt * 32 + ((i < 8) ? (((lane >> 4) << 3) + i) : (16 + ((lane >> 4) << 3) + (i - 8)));
  int n = nt * 16 + (lane & 15);
  int cin = Kreal / taps;
  int tap = k / cin, ci = k % cin;
  int ksrc = ci * taps + tap;
  float v = (n < Nreal) ? src[(size_t)n * Kreal + ksrc] : 0.f;
  dst[idx] = f2bf(v);
}

// ---------------------------------------------------------------- WMMA GEMM
enum { AM_ROW = 0, AM_IM2COL = 1, AM_DEF = 2, AM_XS = 3 };

// One wave computes a (MT*16)-row x (NT*16)-col strip. Per K-step: MT A
// fragments are gathered once and reused across NT B fragments -> MT*NT
// back-to-back v_wmma_f32_16x16x32_bf16, each B fragment reused MT times.
template <int AMODE, int NT, int MT>
__global__ void k_gemm(const float* __restrict__ A, const __bf16* __restrict__ Wf,
                       const float* __restrict__ bias, float* __restrict__ out,
                       int Kdim, int Npad, int Nreal, int omode, int Cin,
                       int total_groups) {
  int lane  = threadIdx.x & 31;
  int group = (int)((blockIdx.x * blockDim.x + threadIdx.x) >> 5);
  if (group >= total_groups) return;
  int ntn = Npad >> 4;
  int ng  = ntn / NT;
  int m0  = (group / ng) * (16 * MT);
  int nt0 = (group % ng) * NT;
  int khalf = (lane >> 4) << 3;

  f32x8 acc[MT][NT];
#pragma unroll
  for (int mt = 0; mt < MT; ++mt)
#pragma unroll
    for (int t = 0; t < NT; ++t)
#pragma unroll
      for (int j = 0; j < 8; ++j) acc[mt][t][j] = 0.f;

  bf16x16 af[MT];
  const __bf16* wf = Wf;

  auto mmstep = [&](int kt) {
#pragma unroll
    for (int t = 0; t < NT; ++t) {
      bf16x16 bfv = *(const bf16x16*)(wf + (((size_t)kt * ntn + nt0 + t) * 32 + lane) * 16);
#pragma unroll
      for (int mt = 0; mt < MT; ++mt)
        acc[mt][t] = __builtin_amdgcn_wmma_f32_16x16x32_bf16(false, af[mt], false, bfv,
                                                             (short)0, acc[mt][t], false, false);
    }
  };

  if (AMODE == AM_ROW) {
    const float* ar[MT];
#pragma unroll
    for (int mt = 0; mt < MT; ++mt)
      ar[mt] = A + (size_t)(m0 + mt * 16 + (lane & 15)) * Kdim + khalf;
    for (int k0 = 0; k0 < Kdim; k0 += 32) {
#pragma unroll
      for (int mt = 0; mt < MT; ++mt) {
        fill8(af[mt], 0, *(const float4*)(ar[mt] + k0),      *(const float4*)(ar[mt] + k0 + 4));
        fill8(af[mt], 8, *(const float4*)(ar[mt] + k0 + 16), *(const float4*)(ar[mt] + k0 + 20));
      }
      mmstep(k0 >> 5);
    }
  } else if (AMODE == AM_XS) {
    // A = xa_c [B][L][192]; rows = (b, dir, l); per-direction weight bank
    wf = Wf + (size_t)((m0 >> 12) & 3) * (size_t)Kdim * Npad;
    const float* ar[MT];
#pragma unroll
    for (int mt = 0; mt < MT; ++mt) {
      int mrow = m0 + mt * 16 + (lane & 15);
      int xb = mrow >> 14, xk = (mrow >> 12) & 3, xl = mrow & 4095;
      int l2;
      if      (xk == 0) l2 = xl;
      else if (xk == 1) l2 = ((xl & 63) << 6) | (xl >> 6);
      else if (xk == 2) l2 = 4095 - xl;
      else { int j = 4095 - xl; l2 = ((j & 63) << 6) | (j >> 6); }
      ar[mt] = A + ((size_t)xb * 4096 + l2) * 192 + khalf;
    }
    for (int k0 = 0; k0 < Kdim; k0 += 32) {
#pragma unroll
      for (int mt = 0; mt < MT; ++mt) {
        fill8(af[mt], 0, *(const float4*)(ar[mt] + k0),      *(const float4*)(ar[mt] + k0 + 4));
        fill8(af[mt], 8, *(const float4*)(ar[mt] + k0 + 16), *(const float4*)(ar[mt] + k0 + 20));
      }
      mmstep(k0 >> 5);
    }
  } else if (AMODE == AM_IM2COL) {
    // A is zero-padded NCHW [B][Cin][66*66]; tap-major K: k = tap*Cin + ci
    int b = m0 >> 12;
    int hh[MT], ww[MT];
#pragma unroll
    for (int mt = 0; mt < MT; ++mt) {
      int hw = (m0 + mt * 16 + (lane & 15)) & 4095;
      hh[mt] = hw >> 6; ww[mt] = hw & 63;
    }
    const float* abase = A + (size_t)b * Cin * PP;
#pragma unroll
    for (int tap = 0; tap < 9; ++tap) {
      const float* ap[MT];
#pragma unroll
      for (int mt = 0; mt < MT; ++mt)
        ap[mt] = abase + (hh[mt] + tap / 3) * 66 + ww[mt] + (tap % 3);
      for (int ci0 = 0; ci0 < Cin; ci0 += 32) {
#pragma unroll
        for (int mt = 0; mt < MT; ++mt)
#pragma unroll
          for (int i = 0; i < 16; ++i) {
            int ci = ci0 + ((i < 8) ? (khalf + i) : (16 + khalf + (i - 8)));
            af[mt][i] = f2bf(ap[mt][(size_t)ci * PP]);
          }
        mmstep((tap * Cin + ci0) >> 5);
      }
    }
  } else { // AM_DEF: deformed [B][3][96][HW]; tap-major K: k = kk*96 + ci
    int b = m0 >> 12;
    int hw[MT];
#pragma unroll
    for (int mt = 0; mt < MT; ++mt)
      hw[mt] = (m0 + mt * 16 + (lane & 15)) & 4095;
#pragma unroll
    for (int kk = 0; kk < 3; ++kk) {
      const float* ap[MT];
#pragma unroll
      for (int mt = 0; mt < MT; ++mt)
        ap[mt] = A + (size_t)((b * 3 + kk) * 96) * 4096 + hw[mt];
      for (int ci0 = 0; ci0 < 96; ci0 += 32) {
#pragma unroll
        for (int mt = 0; mt < MT; ++mt)
#pragma unroll
          for (int i = 0; i < 16; ++i) {
            int ci = ci0 + ((i < 8) ? (khalf + i) : (16 + khalf + (i - 8)));
            af[mt][i] = f2bf(ap[mt][(size_t)ci * 4096]);
          }
        mmstep((kk * 96 + ci0) >> 5);
      }
    }
  }

  // -------- store (C layout: lane holds col n=lane%16, vgpr r -> m=(lane/16)*8+r)
#pragma unroll
  for (int mt = 0; mt < MT; ++mt)
#pragma unroll
    for (int t = 0; t < NT; ++t) {
      int ncol = (nt0 + t) * 16 + (lane & 15);
      if (ncol >= Nreal) continue;
      float bv = bias ? bias[ncol] : 0.f;
#pragma unroll
      for (int r = 0; r < 8; ++r) {
        int mr = m0 + mt * 16 + ((lane >> 4) << 3) + r;
        float v = acc[mt][t][r] + bv;
        if (omode == 0) out[(size_t)mr * Npad + ncol] = v;                      // row-major
        else out[(size_t)((mr >> 12) * Nreal + ncol) * 4096 + (mr & 4095)] = v; // NCHW
      }
    }
}

// ---------------------------------------------------------------- zero-padded copy
__global__ void k_padcpy(const float* __restrict__ src, float* __restrict__ dst,
                         long long tot) {
  long long idx = (long long)blockIdx.x * 256 + threadIdx.x;
  if (idx >= tot) return;
  int p = (int)(idx % PP); long long bc = idx / PP;
  int py = p / 66, px = p % 66;
  float v = 0.f;
  if (py >= 1 && py <= 64 && px >= 1 && px <= 64)
    v = src[bc * 4096 + (py - 1) * 64 + (px - 1)];
  dst[idx] = v;
}

__global__ void k_fill0(float* __restrict__ p, long long n) {
  long long idx = (long long)blockIdx.x * 256 + threadIdx.x;
  if (idx < n) p[idx] = 0.f;
}

// ---------------------------------------------------------------- offset conv 96->6
__global__ void k_offconv(const float* __restrict__ in, const float* __restrict__ w,
                          const float* __restrict__ bi, float* __restrict__ out) {
  int i = blockIdx.x * 256 + threadIdx.x;
  if (i >= Bn * 6 * HWn) return;
  int hw = i & 4095; int o = (i >> 12) % 6; int b = i / (6 * 4096);
  int hh = hw >> 6, ww = hw & 63;
  float s = bi[o];
  for (int ci = 0; ci < 96; ++ci)
    for (int dy = -1; dy <= 1; ++dy) {
      int y = hh + dy; if ((unsigned)y >= 64u) continue;
      for (int dx = -1; dx <= 1; ++dx) {
        int x = ww + dx; if ((unsigned)x >= 64u) continue;
        s += w[((o * 96 + ci) * 3 + (dy + 1)) * 3 + (dx + 1)] *
             in[(size_t)(b * 96 + ci) * 4096 + (y << 6) + x];
      }
    }
  out[i] = s;
}

__global__ void k_bnstats(const float* __restrict__ off, float* __restrict__ mo,
                          float* __restrict__ ro) {
  int o = blockIdx.x; int t = threadIdx.x;
  __shared__ float ssum[256], ssq[256];
  float s = 0.f, q = 0.f;
  for (int b = 0; b < Bn; ++b)
    for (int i = t; i < 4096; i += 256) {
      float v = off[(size_t)(b * 6 + o) * 4096 + i]; s += v; q += v * v;
    }
  ssum[t] = s; ssq[t] = q; __syncthreads();
  for (int st = 128; st > 0; st >>= 1) {
    if (t < st) { ssum[t] += ssum[t + st]; ssq[t] += ssq[t + st]; }
    __syncthreads();
  }
  if (t == 0) {
    float m = ssum[0] / (float)(Bn * 4096);
    float v = ssq[0] / (float)(Bn * 4096) - m * m;
    mo[o] = m; ro[o] = rsqrtf(v + 1e-5f);
  }
}

__global__ void k_bntanh(float* __restrict__ off, const float* mo, const float* ro,
                         const float* g, const float* bb) {
  int i = blockIdx.x * 256 + threadIdx.x;
  if (i >= Bn * 6 * HWn) return;
  int o = (i >> 12) % 6;
  off[i] = tanhf((off[i] - mo[o]) * ro[o] * g[o] + bb[o]);
}

// ---------------------------------------------------------------- bilinear sampling
// writes deformed [B][3][96][HW] (tap-major for the WMMA K-order)
__global__ void k_sample(const float* __restrict__ f, const float* __restrict__ off,
                         float* __restrict__ def, int morph) {
  int i = blockIdx.x * 256 + threadIdx.x;
  if (i >= Bn * 3 * 96 * HWn) return;
  int hw = i & 4095; int q = i >> 12;
  int c = q % 96; int k = (q / 96) % 3; int b = q / 288;
  int hh = hw >> 6, ww = hw & 63;
  float yy, xx;
  if (morph == 0) {
    float dy = (k == 1) ? 0.f : off[(size_t)(b * 6 + k) * 4096 + hw];     // yoff
    yy = hh + dy; xx = (float)(ww + (k - 1));
  } else {
    float dx = (k == 1) ? 0.f : off[(size_t)(b * 6 + 3 + k) * 4096 + hw]; // xoff
    xx = ww + dx; yy = (float)(hh + (k - 1));
  }
  int y0 = (int)floorf(yy), x0 = (int)floorf(xx);
  int y0i = min(max(y0, 0), 63), y1i = min(max(y0 + 1, 0), 63);
  int x0i = min(max(x0, 0), 63), x1i = min(max(x0 + 1, 0), 63);
  const float* fb = f + (size_t)(b * 96 + c) * 4096;
  float va = fb[(y0i << 6) + x0i], vb = fb[(y0i << 6) + x1i];
  float vc = fb[(y1i << 6) + x0i], vd = fb[(y1i << 6) + x1i];
  float y0f = (float)y0i, y1f = (float)y1i, x0f = (float)x0i, x1f = (float)x1i;
  float wa = (y1f - yy) * (x1f - xx), wb2 = (y1f - yy) * (xx - x0f);
  float wc = (yy - y0f) * (x1f - xx), wd  = (yy - y0f) * (xx - x0f);
  def[i] = va * wa + vb * wb2 + vc * wc + vd * wd;
}

// ---------------------------------------------------------------- group norm
__global__ void k_gnstats(const float* __restrict__ x, float* __restrict__ mo,
                          float* __restrict__ ro) {
  int bg = blockIdx.x; int b = bg / 24, g = bg % 24; int t = threadIdx.x;
  __shared__ float ssum[256], ssq[256];
  float s = 0.f, q = 0.f;
  for (int c = 0; c < 4; ++c) {
    const float* p = x + (size_t)(b * 96 + g * 4 + c) * 4096;
    for (int i = t; i < 4096; i += 256) { float v = p[i]; s += v; q += v * v; }
  }
  ssum[t] = s; ssq[t] = q; __syncthreads();
  for (int st = 128; st > 0; st >>= 1) {
    if (t < st) { ssum[t] += ssum[t + st]; ssq[t] += ssq[t + st]; }
    __syncthreads();
  }
  if (t == 0) {
    float m = ssum[0] / 16384.f;
    float v = ssq[0] / 16384.f - m * m;
    mo[bg] = m; ro[bg] = rsqrtf(v + 1e-5f);
  }
}

// dst padded (66x66, border untouched) when padded=1, else dense NCHW 4096
__global__ void k_gnapply(const float* __restrict__ x, const float* mo, const float* ro,
                          const float* g, const float* bb, float* __restrict__ dst,
                          int dstC, int c_off, int padded) {
  int i = blockIdx.x * 256 + threadIdx.x;
  if (i >= Bn * 96 * HWn) return;
  int hw = i & 4095; int c = (i >> 12) % 96; int b = i / (96 * 4096);
  int bg = b * 24 + (c >> 2);
  float v = (x[i] - mo[bg]) * ro[bg] * g[c] + bb[c];
  v = fmaxf(v, 0.f);
  if (padded)
    dst[(size_t)(b * dstC + c_off + c) * PP + ((hw >> 6) + 1) * 66 + (hw & 63) + 1] = v;
  else
    dst[(size_t)(b * dstC + c_off + c) * 4096 + hw] = v;
}

// ---------------------------------------------------------------- layer norm (C=96)
__global__ void k_ln1(const float* __restrict__ in, const float* g, const float* bb,
                      float* __restrict__ xn) {
  int i = blockIdx.x * 256 + threadIdx.x;
  if (i >= Bn * Ln) return;
  int b = i >> 12, l = i & 4095;
  const float* p = in + (size_t)b * 96 * 4096 + l;
  float s = 0.f;
  for (int c = 0; c < 96; ++c) s += p[(size_t)c * 4096];
  float m = s / 96.f;
  float q = 0.f;
  for (int c = 0; c < 96; ++c) { float d = p[(size_t)c * 4096] - m; q += d * d; }
  float rs = rsqrtf(q / 96.f + 1e-6f);
  float* o = xn + (size_t)i * 96;
  for (int c = 0; c < 96; ++c) o[c] = (p[(size_t)c * 4096] - m) * rs * g[c] + bb[c];
}

// ---------------------------------------------------------------- depthwise conv + silu
// out xac layout [B][L][192]
__global__ void k_dwconv(const float* __restrict__ xz, const float* __restrict__ w,
                         const float* __restrict__ bi, float* __restrict__ xac) {
  int i = blockIdx.x * 256 + threadIdx.x;
  if (i >= Bn * DIn * HWn) return;
  int d = i % 192; int hw = (i / 192) & 4095; int b = i / (192 * 4096);
  int hh = hw >> 6, ww = hw & 63;
  float s = bi[d];
  for (int dy = -1; dy <= 1; ++dy) {
    int y = hh + dy; if ((unsigned)y >= 64u) continue;
    for (int dx = -1; dx <= 1; ++dx) {
      int x = ww + dx; if ((unsigned)x >= 64u) continue;
      s += w[(d * 3 + (dy + 1)) * 3 + (dx + 1)] *
           xz[(size_t)(b * 4096 + (y << 6) + x) * 384 + d];
    }
  }
  float sig = 1.f / (1.f + expf(-s));
  xac[(size_t)(b * 4096 + hw) * 192 + d] = s * sig;
}

// ---------------------------------------------------------------- dt proj + softplus
__global__ void k_dt(const float* __restrict__ xdbl, const float* __restrict__ dpw,
                     const float* __restrict__ dpb, float* __restrict__ dt) {
  size_t i = (size_t)blockIdx.x * 256 + threadIdx.x;
  if (i >= (size_t)Bn * Kd * Ln * 192) return;
  int d = (int)(i % 192); size_t row = i / 192;   // row = (b*K+k)*L + l
  int kdir = (int)((row >> 12) & 3);
  const float* xr = xdbl + row * 48;
  const float* wr = dpw + (size_t)(kdir * 192 + d) * 6;
  float s = dpb[kdir * 192 + d];
#pragma unroll
  for (int r = 0; r < 6; ++r) s += xr[r] * wr[r];
  dt[i] = (s > 20.f) ? s : log1pf(expf(s));
}

// ---------------------------------------------------------------- selective scan
// grid = B*K blocks, 192 threads (6 wave32). dty: dt in, out_y (y + Ds*u) out.
// Per-step B/C (32 floats) staged to LDS via gfx1250 async global->LDS copy
// (ASYNCcnt) when the toolchain exposes the builtin.
__global__ void k_scan(const float* __restrict__ xac, const float* __restrict__ xdbl,
                       float* __restrict__ dty, const float* __restrict__ alog,
                       const float* __restrict__ Ds) {
  int bk = blockIdx.x;
  int b = bk >> 2, kdir = bk & 3;
  int d = threadIdx.x;
  float Av[16];
#pragma unroll
  for (int n = 0; n < 16; ++n) Av[n] = -expf(alog[(size_t)(kdir * 192 + d) * 16 + n]);
  float Dv = Ds[kdir * 192 + d];
  float h[16];
#pragma unroll
  for (int n = 0; n < 16; ++n) h[n] = 0.f;
  __shared__ float sBC[32];
  size_t rowbase = (size_t)bk * 4096;
  for (int l = 0; l < 4096; ++l) {
    size_t row = rowbase + l;
    if (d < 32) {
#if HAVE_ASYNC_LDS
      __builtin_amdgcn_global_load_async_to_lds_b32(
          (__attribute__((address_space(1))) int*)(xdbl + row * 48 + 6 + d),
          (__attribute__((address_space(3))) int*)&sBC[d], 0, 0);
#else
      sBC[d] = xdbl[row * 48 + 6 + d];
#endif
    }
#if HAVE_ASYNC_LDS
    __builtin_amdgcn_s_wait_asynccnt(0);
#endif
    __syncthreads();
    float dtv = dty[row * 192 + d];
    if (l + 4 < 4096) __builtin_prefetch(&dty[(row + 4) * 192 + d], 0, 1);
    int l2;
    if      (kdir == 0) l2 = l;
    else if (kdir == 1) l2 = ((l & 63) << 6) | (l >> 6);
    else if (kdir == 2) l2 = 4095 - l;
    else { int j = 4095 - l; l2 = ((j & 63) << 6) | (j >> 6); }
    float u = xac[(size_t)(b * 4096 + l2) * 192 + d];
    float du = dtv * u;
    float y = 0.f;
#pragma unroll
    for (int n = 0; n < 16; ++n) {
      h[n] = h[n] * __expf(dtv * Av[n]) + du * sBC[n];
      y += h[n] * sBC[16 + n];
    }
    __syncthreads();
    dty[row * 192 + d] = y + Dv * u;
  }
}

// ---------------------------------------------------------------- direction combine
__global__ void k_combine(const float* __restrict__ yout, float* __restrict__ ybuf) {
  size_t i = (size_t)blockIdx.x * 256 + threadIdx.x;
  if (i >= (size_t)Bn * Ln * 192) return;
  int d = (int)(i % 192); size_t bl = i / 192;
  int b = (int)(bl >> 12); int l = (int)(bl & 4095);
  int lt = ((l & 63) << 6) | (l >> 6);
  size_t base = (size_t)b * 4 * 4096 * 192;
  float v = yout[base + (size_t)(0 * 4096 + l) * 192 + d]
          + yout[base + (size_t)(1 * 4096 + lt) * 192 + d]
          + yout[base + (size_t)(2 * 4096 + (4095 - l)) * 192 + d]
          + yout[base + (size_t)(3 * 4096 + (4095 - lt)) * 192 + d];
  ybuf[i] = v;
}

// ---------------------------------------------------------------- out-norm + silu gate
__global__ void k_outnorm(float* __restrict__ ybuf, const float* __restrict__ xz,
                          const float* g, const float* bb) {
  int i = blockIdx.x * 256 + threadIdx.x;
  if (i >= Bn * Ln) return;
  float* p = ybuf + (size_t)i * 192;
  const float* zp = xz + (size_t)i * 384 + 192;
  float s = 0.f;
  for (int d = 0; d < 192; ++d) s += p[d];
  float m = s / 192.f;
  float q = 0.f;
  for (int d = 0; d < 192; ++d) { float t = p[d] - m; q += t * t; }
  float rs = rsqrtf(q / 192.f + 1e-5f);
  for (int d = 0; d < 192; ++d) {
    float z = zp[d]; float sil = z / (1.f + expf(-z));
    p[d] = ((p[d] - m) * rs * g[d] + bb[d]) * sil;
  }
}

// ---------------------------------------------------------------- residual -> padded x2
__global__ void k_x2pad(const float* __restrict__ in, const float* __restrict__ attn,
                        const float* __restrict__ x1, float* __restrict__ out) {
  long long idx = (long long)blockIdx.x * 256 + threadIdx.x;
  if (idx >= (long long)Bn * 96 * PP) return;
  int p = (int)(idx % PP); int c = (int)((idx / PP) % 96); int b = (int)(idx / ((long long)96 * PP));
  int py = p / 66, px = p % 66;
  float v = 0.f;
  if (py >= 1 && py <= 64 && px >= 1 && px <= 64) {
    int hw = (py - 1) * 64 + (px - 1);
    size_t j = (size_t)(b * 96 + c) * 4096 + hw;
    v = in[j] + x1[j] + attn[((size_t)(b * 4096 + hw)) * 96 + c];
  }
  out[idx] = v;
}

// ================================================================ host
static inline int ceil_div(long long a, long long b) { return (int)((a + b - 1) / b); }

extern "C" void kernel_launch(void* const* d_in, const int* in_sizes, int n_in,
                              void* d_out, int out_size, void* d_ws, size_t ws_size,
                              hipStream_t stream) {
  const float* input     = (const float*)d_in[0];
  const float* ln1_g     = (const float*)d_in[1];
  const float* ln1_b     = (const float*)d_in[2];
  const float* in_proj_w = (const float*)d_in[3];
  const float* conv_dw_w = (const float*)d_in[4];
  const float* conv_dw_b = (const float*)d_in[5];
  const float* x_proj_w  = (const float*)d_in[6];
  const float* dt_proj_w = (const float*)d_in[7];
  const float* dt_proj_b = (const float*)d_in[8];
  const float* A_logs    = (const float*)d_in[9];
  const float* Ds        = (const float*)d_in[10];
  const float* out_n_g   = (const float*)d_in[11];
  const float* out_n_b   = (const float*)d_in[12];
  const float* out_proj_w= (const float*)d_in[13];
  const float* c00_w     = (const float*)d_in[14];
  const float* c00_b     = (const float*)d_in[15];
  const float* c00_gn_g  = (const float*)d_in[16];
  const float* c00_gn_b  = (const float*)d_in[17];
  const float* ox_ow     = (const float*)d_in[18];
  const float* ox_ob     = (const float*)d_in[19];
  const float* ox_bn_g   = (const float*)d_in[20];
  const float* ox_bn_b   = (const float*)d_in[21];
  const float* ox_dw     = (const float*)d_in[22];
  const float* ox_db     = (const float*)d_in[23];
  const float* ox_gn_g   = (const float*)d_in[24];
  const float* ox_gn_b   = (const float*)d_in[25];
  const float* oy_ow     = (const float*)d_in[26];
  const float* oy_ob     = (const float*)d_in[27];
  const float* oy_bn_g   = (const float*)d_in[28];
  const float* oy_bn_b   = (const float*)d_in[29];
  const float* oy_dw     = (const float*)d_in[30];
  const float* oy_db     = (const float*)d_in[31];
  const float* oy_gn_g   = (const float*)d_in[32];
  const float* oy_gn_b   = (const float*)d_in[33];
  const float* c1_w      = (const float*)d_in[34];
  const float* c1_b      = (const float*)d_in[35];
  const float* c1_gn_g   = (const float*)d_in[36];
  const float* c1_gn_b   = (const float*)d_in[37];
  const float* c2_w      = (const float*)d_in[38];
  const float* c2_b      = (const float*)d_in[39];

  char* ws = (char*)d_ws;
  size_t off = 0;
  auto alloc = [&](size_t bytes) -> char* {
    char* p = ws + off;
    off += (bytes + 255) & ~(size_t)255;
    return p;
  };

  // bf16 weight fragment banks
  __bf16* wpk_c00  = (__bf16*)alloc((size_t)864  * 96  * 2);
  __bf16* wpk_oxdw = (__bf16*)alloc((size_t)288  * 96  * 2);
  __bf16* wpk_oydw = (__bf16*)alloc((size_t)288  * 96  * 2);
  __bf16* wpk_c1   = (__bf16*)alloc((size_t)2592 * 96  * 2);
  __bf16* wpk_c2   = (__bf16*)alloc((size_t)864  * 96  * 2);
  __bf16* wpk_inp  = (__bf16*)alloc((size_t)96   * 384 * 2);
  __bf16* wpk_xp   = (__bf16*)alloc((size_t)4 * 192 * 48 * 2);
  __bf16* wpk_outp = (__bf16*)alloc((size_t)192  * 96  * 2);

  // activations / scratch
  float* inpad = (float*)alloc((size_t)Bn * 96  * PP  * 4);  // zero-padded input
  float* tmp   = (float*)alloc((size_t)Bn * 96  * HWn * 4);  // conv pre-GN (NCHW)
  float* xcatp = (float*)alloc((size_t)Bn * 288 * PP  * 4);  // padded [x00|x0x|x0y]
  float* x1b   = (float*)alloc((size_t)Bn * 96  * HWn * 4);
  float* offb  = (float*)alloc((size_t)Bn * 6   * HWn * 4);
  float* bn_m  = (float*)alloc(64 * 4);
  float* bn_r  = (float*)alloc(64 * 4);
  float* gn_m  = (float*)alloc((size_t)Bn * 24 * 4);
  float* gn_r  = (float*)alloc((size_t)Bn * 24 * 4);
  float* defb  = (float*)alloc((size_t)Bn * 3 * 96 * HWn * 4);
  float* xnb   = (float*)alloc((size_t)Bn * Ln * 96  * 4);
  float* xzb   = (float*)alloc((size_t)Bn * Ln * 384 * 4);
  float* xac   = (float*)alloc((size_t)Bn * Ln * 192 * 4);   // [B][L][192]
  float* xdbl  = (float*)alloc((size_t)Bn * 4 * Ln * 48 * 4);
  float* dty   = (float*)alloc((size_t)Bn * 4 * Ln * 192 * 4);
  float* ybuf  = (float*)alloc((size_t)Bn * Ln * 192 * 4);
  float* attn  = (float*)alloc((size_t)Bn * Ln * 96  * 4);
  float* x2p   = (float*)alloc((size_t)Bn * 96 * PP * 4);    // padded residual
  (void)ws_size; (void)in_sizes; (void)n_in; (void)out_size;

  // ---- pack weights into fragment banks
  auto packN = [&](const float* src, __bf16* dst, int Kr, int Nr, int Np, int taps) {
    long long tot = (long long)Kr * Np;
    k_pack<<<ceil_div(tot, 256), 256, 0, stream>>>(src, dst, Kr, Nr, Np, taps);
  };
  packN(c00_w,      wpk_c00,  864,  96,  96, 9);
  packN(ox_dw,      wpk_oxdw, 288,  96,  96, 3);
  packN(oy_dw,      wpk_oydw, 288,  96,  96, 3);
  packN(c1_w,       wpk_c1,  2592,  96,  96, 9);
  packN(c2_w,       wpk_c2,   864,  96,  96, 9);
  packN(in_proj_w,  wpk_inp,   96, 384, 384, 1);
  for (int k = 0; k < 4; ++k)
    packN(x_proj_w + (size_t)k * 38 * 192, wpk_xp + (size_t)k * 192 * 48,
          192, 38, 48, 1);
  packN(out_proj_w, wpk_outp, 192,  96,  96, 1);

  const int EW96 = Bn * 96 * HWn;
  auto launch_gemm = [&](auto kern, const float* A, const __bf16* Wf, const float* bias,
                         float* out, int M, int Kdim, int Npad, int Nreal, int omode,
                         int Cin, int NT, int MT) {
    int groups = (M / (16 * MT)) * ((Npad / 16) / NT);
    kern<<<ceil_div((long long)groups * 32, 256), 256, 0, stream>>>(
        A, Wf, bias, out, Kdim, Npad, Nreal, omode, Cin, groups);
  };

  // ---- pad input, zero xcat borders
  k_padcpy<<<ceil_div((long long)Bn * 96 * PP, 256), 256, 0, stream>>>(
      input, inpad, (long long)Bn * 96 * PP);
  k_fill0<<<ceil_div((long long)Bn * 288 * PP, 256), 256, 0, stream>>>(
      xcatp, (long long)Bn * 288 * PP);

  // ---- c00: conv3x3 96->96 + GN + ReLU -> xcatp[:, 0:96]
  launch_gemm(k_gemm<AM_IM2COL, 6, 2>, inpad, wpk_c00, c00_b, tmp, 16384, 864, 96, 96, 1, 96, 6, 2);
  k_gnstats<<<Bn * 24, 256, 0, stream>>>(tmp, gn_m, gn_r);
  k_gnapply<<<ceil_div(EW96, 256), 256, 0, stream>>>(tmp, gn_m, gn_r, c00_gn_g,
                                                     c00_gn_b, xcatp, 288, 0, 1);
  // ---- dsconv branches
  for (int morph = 0; morph < 2; ++morph) {
    const float* ow  = morph ? oy_ow   : ox_ow;
    const float* ob  = morph ? oy_ob   : ox_ob;
    const float* bng = morph ? oy_bn_g : ox_bn_g;
    const float* bnb = morph ? oy_bn_b : ox_bn_b;
    const float* db  = morph ? oy_db   : ox_db;
    const float* gg  = morph ? oy_gn_g : ox_gn_g;
    const float* gb  = morph ? oy_gn_b : ox_gn_b;
    __bf16* wpk      = morph ? wpk_oydw : wpk_oxdw;
    k_offconv<<<ceil_div(Bn * 6 * HWn, 256), 256, 0, stream>>>(input, ow, ob, offb);
    k_bnstats<<<6, 256, 0, stream>>>(offb, bn_m, bn_r);
    k_bntanh<<<ceil_div(Bn * 6 * HWn, 256), 256, 0, stream>>>(offb, bn_m, bn_r, bng, bnb);
    k_sample<<<ceil_div(Bn * 3 * 96 * HWn, 256), 256, 0, stream>>>(input, offb, defb, morph);
    launch_gemm(k_gemm<AM_DEF, 6, 2>, defb, wpk, db, tmp, 16384, 288, 96, 96, 1, 96, 6, 2);
    k_gnstats<<<Bn * 24, 256, 0, stream>>>(tmp, gn_m, gn_r);
    k_gnapply<<<ceil_div(EW96, 256), 256, 0, stream>>>(tmp, gn_m, gn_r, gg, gb,
                                                       xcatp, 288, 96 + morph * 96, 1);
  }
  // ---- c1: conv3x3 288->96 + GN + ReLU -> x1 (dense)
  launch_gemm(k_gemm<AM_IM2COL, 6, 2>, xcatp, wpk_c1, c1_b, tmp, 16384, 2592, 96, 96, 1, 288, 6, 2);
  k_gnstats<<<Bn * 24, 256, 0, stream>>>(tmp, gn_m, gn_r);
  k_gnapply<<<ceil_div(EW96, 256), 256, 0, stream>>>(tmp, gn_m, gn_r, c1_gn_g,
                                                     c1_gn_b, x1b, 96, 0, 0);
  // ---- ss2d
  k_ln1<<<ceil_div(Bn * Ln, 256), 256, 0, stream>>>(input, ln1_g, ln1_b, xnb);
  launch_gemm(k_gemm<AM_ROW, 6, 2>, xnb, wpk_inp, nullptr, xzb, 16384, 96, 384, 384, 0, 0, 6, 2);
  k_dwconv<<<ceil_div(Bn * DIn * HWn, 256), 256, 0, stream>>>(xzb, conv_dw_w, conv_dw_b, xac);
  launch_gemm(k_gemm<AM_XS, 3, 2>, xac, wpk_xp, nullptr, xdbl, 65536, 192, 48, 38, 0, 0, 3, 2);
  {
    long long tot = (long long)Bn * Kd * Ln * 192;
    k_dt<<<ceil_div(tot, 256), 256, 0, stream>>>(xdbl, dt_proj_w, dt_proj_b, dty);
  }
  k_scan<<<Bn * Kd, 192, 0, stream>>>(xac, xdbl, dty, A_logs, Ds);
  {
    long long tot = (long long)Bn * Ln * 192;
    k_combine<<<ceil_div(tot, 256), 256, 0, stream>>>(dty, ybuf);
  }
  k_outnorm<<<ceil_div(Bn * Ln, 256), 256, 0, stream>>>(ybuf, xzb, out_n_g, out_n_b);
  launch_gemm(k_gemm<AM_ROW, 6, 2>, ybuf, wpk_outp, nullptr, attn, 16384, 192, 96, 96, 0, 0, 6, 2);
  // ---- residual (padded) + final conv -> d_out
  k_x2pad<<<ceil_div((long long)Bn * 96 * PP, 256), 256, 0, stream>>>(input, attn, x1b, x2p);
  launch_gemm(k_gemm<AM_IM2COL, 6, 2>, x2p, wpk_c2, c2_b, (float*)d_out, 16384, 864, 96, 96, 1, 96, 6, 2);
}
